// SimpleAttention_71502615544618
// MI455X (gfx1250) — matrix operand order repaired and checked by reading, not compiled, not run on hardware
//
#include <hip/hip_runtime.h>
#include <hip/hip_bf16.h>
#include <stdint.h>

// ---------------------------------------------------------------------------
// CDNA5 (gfx1250) fused causal attention, bf16 WMMA path.
//   B=2, T=2048, C=1024, H=16, D=64
//   Pipeline: cvt -> qkv gemm (64x64 register-blocked, V stored transposed)
//             -> flash attention (32-row query tiles) -> out-proj gemm
// ---------------------------------------------------------------------------

typedef __attribute__((ext_vector_type(16))) __bf16        v16bf;
typedef __attribute__((ext_vector_type(8)))  float         v8f;
typedef __attribute__((ext_vector_type(4)))  unsigned int  v4u;

union FragB {
    v16bf          v;
    v4u            q[2];
    unsigned short h[16];
};

__device__ __forceinline__ unsigned short f32_to_bf16(float f) {
    uint32_t u = __builtin_bit_cast(uint32_t, f);
    u += ((u >> 16) & 1u) + 0x7FFFu;   // round to nearest even
    return (unsigned short)(u >> 16);
}

__device__ __forceinline__ v8f wmma_bf16(v16bf a, v16bf b, v8f c) {
    return __builtin_amdgcn_wmma_f32_16x16x32_bf16(
        /*neg_a=*/false, a, /*neg_b=*/false, b,
        /*c_mod=*/(short)0, c, /*reuse_a=*/false, /*reuse_b=*/false);
}

// A fragment of a 16x32 (MxK) tile, row-major source with leading dim `ld`.
// ISA layout: lanes 0-15 row M=lane hold K chunks [0..7],[16..23];
//             lanes 16-31 row M=lane-16 hold K chunks [8..15],[24..31].
__device__ __forceinline__ v16bf load_frag_a(const unsigned short* p, int ld) {
    const int lane = threadIdx.x & 31;
    const int r    = lane & 15;
    const int half = lane >> 4;
    FragB f;
    const unsigned short* row = p + (size_t)r * ld;
    f.q[0] = *(const v4u*)(row + 8 * half);
    f.q[1] = *(const v4u*)(row + 16 + 8 * half);
    return f.v;
}

// B fragment of a 32x16 (KxN) tile where B[k][n] = src[n*ld + k]
// (N walks rows of the source; K contiguous).  Lane n (0-15) holds K=[0..7],[16..23],
// lane n+16 holds K=[8..15],[24..31].
__device__ __forceinline__ v16bf load_frag_b_rowN(const unsigned short* p, int ld) {
    const int lane = threadIdx.x & 31;
    const int n    = lane & 15;
    const int half = lane >> 4;
    FragB f;
    const unsigned short* row = p + (size_t)n * ld + 8 * half;
    f.q[0] = *(const v4u*)(row);
    f.q[1] = *(const v4u*)(row + 16);
    return f.v;
}

// ---------------------------------------------------------------------------
// Kernel 1: fp32 -> bf16 cast
// ---------------------------------------------------------------------------
__global__ __launch_bounds__(256)
void cvt_f32_bf16(const float* __restrict__ in, unsigned short* __restrict__ out, int n) {
    int i = blockIdx.x * blockDim.x + threadIdx.x;
    if (i < n) out[i] = f32_to_bf16(in[i]);
}

// ---------------------------------------------------------------------------
// Kernel 2: qkv = x @ w_qkv^T    (A=[4096,1024] bf16, B=[3072,1024] bf16)
// 64x64 register-blocked: each wave owns a 4x4 grid of 16x16 WMMA tiles,
// so every A/B fragment fetched from L2 feeds 4 WMMAs.
// Q,K scatter into [2][B][H][T][D]; V scatters TRANSPOSED into [B][H][D][T]
// so the attention PV B-fragments are contiguous b128 loads.
// ---------------------------------------------------------------------------
__global__ __launch_bounds__(256)
void gemm_qkv(const unsigned short* __restrict__ xb,
              const unsigned short* __restrict__ wb,
              unsigned short* __restrict__ qk,    // [2][2][16][2048][64]
              unsigned short* __restrict__ vt) {  // [2][16][64][2048]
    const int NMAC = 3072 / 64;                    // 48 macro N tiles
    const int wave = (blockIdx.x * blockDim.x + threadIdx.x) >> 5;
    const int m0   = (wave / NMAC) * 64;
    const int n0   = (wave % NMAC) * 64;

    v8f c[4][4] = {};
    for (int k = 0; k < 1024; k += 32) {
        v16bf a[4], b[4];
#pragma unroll
        for (int i = 0; i < 4; ++i)
            a[i] = load_frag_a(xb + (size_t)(m0 + i * 16) * 1024 + k, 1024);
#pragma unroll
        for (int j = 0; j < 4; ++j)
            b[j] = load_frag_b_rowN(wb + (size_t)(n0 + j * 16) * 1024 + k, 1024);
#pragma unroll
        for (int i = 0; i < 4; ++i)
#pragma unroll
            for (int j = 0; j < 4; ++j)
                c[i][j] = wmma_bf16(a[i], b[j], c[i][j]);
    }

    const int lane = threadIdx.x & 31;
    const int col  = lane & 15;
    const int half = lane >> 4;
#pragma unroll
    for (int i = 0; i < 4; ++i)
#pragma unroll
        for (int j = 0; j < 4; ++j)
#pragma unroll
            for (int v = 0; v < 8; ++v) {
                int row   = m0 + i * 16 + v + 8 * half;   // token in [0,4096)
                int f     = n0 + j * 16 + col;            // feature in [0,3072)
                int which = f >> 10;                      // 0:q 1:k 2:v
                int rem   = f & 1023;
                int h     = rem >> 6;
                int d     = rem & 63;
                int b_    = row >> 11;
                int t     = row & 2047;
                unsigned short val = f32_to_bf16(c[i][j][v]);
                if (which < 2) {
                    size_t idx = ((((size_t)which * 2 + b_) * 16 + h) * 2048 + t) * 64 + d;
                    qk[idx] = val;
                } else {
                    size_t idx = (((size_t)b_ * 16 + h) * 64 + d) * 2048 + t;
                    vt[idx] = val;
                }
            }
}

// ---------------------------------------------------------------------------
// Kernel 3: flash attention.  One wave per (b, h, 32-row query tile):
// two 16-row M tiles share every K and V fragment (2x reuse).
// Keys processed 32 at a time: 8 WMMAs for QK^T, online softmax with
// 16-lane shuffles, P staged through LDS (s_wait_dscnt for the intra-wave
// DS RAW), 8 WMMAs for PV against transposed V.
// ---------------------------------------------------------------------------
__global__ __launch_bounds__(256)
void flash_attn(const unsigned short* __restrict__ qk,  // [2][2][16][2048][64]
                const unsigned short* __restrict__ vt,  // [2][16][64][2048]
                unsigned short* __restrict__ ao) {      // [2][2048][1024]
    __shared__ __align__(16) unsigned short pbuf[8][2][16 * 32];

    const int lane = threadIdx.x & 31;
    const int wid  = threadIdx.x >> 5;
    const int gw   = blockIdx.x * 8 + wid;   // 0..2047
    const int qt   = gw & 63;                // 32-row query tile
    const int bh   = gw >> 6;                // 0..31
    const int b_   = bh >> 4;
    const int h    = bh & 15;
    const size_t plane = (size_t)2048 * 64;

    const unsigned short* qp  = qk + ((size_t)(0 * 2 + b_) * 16 + h) * plane;
    const unsigned short* kp  = qk + ((size_t)(1 * 2 + b_) * 16 + h) * plane;
    const unsigned short* vtp = vt + ((size_t)b_ * 16 + h) * plane; // [64][2048]
    const int q0 = qt * 32;

    // Q fragments: 2 M tiles x 2 K chunks, loaded once.
    v16bf qa[2][2];
#pragma unroll
    for (int mt = 0; mt < 2; ++mt) {
        qa[mt][0] = load_frag_a(qp + (size_t)(q0 + mt * 16) * 64 + 0, 64);
        qa[mt][1] = load_frag_a(qp + (size_t)(q0 + mt * 16) * 64 + 32, 64);
    }

    v8f   oc[2][4] = {};
    float m[2][8], l[2][8];
#pragma unroll
    for (int mt = 0; mt < 2; ++mt)
#pragma unroll
        for (int v = 0; v < 8; ++v) { m[mt][v] = -3.0e38f; l[mt][v] = 0.0f; }

    const int   col   = lane & 15;
    const int   half  = lane >> 4;
    const float scale = 0.125f;              // 1/sqrt(64)

    for (int s0 = 0; s0 < q0 + 32; s0 += 32) {
        __builtin_prefetch(kp + (size_t)(s0 + 32) * 64, 0, 0);

        // ---- K fragments for keys [s0, s0+32), shared by both M tiles ----
        v16bf kb[4];
        kb[0] = load_frag_b_rowN(kp + (size_t)s0 * 64 + 0, 64);
        kb[1] = load_frag_b_rowN(kp + (size_t)s0 * 64 + 32, 64);
        kb[2] = load_frag_b_rowN(kp + (size_t)(s0 + 16) * 64 + 0, 64);
        kb[3] = load_frag_b_rowN(kp + (size_t)(s0 + 16) * 64 + 32, 64);

#pragma unroll
        for (int mt = 0; mt < 2; ++mt) {
            v8f sc0 = {}, sc1 = {};
            sc0 = wmma_bf16(qa[mt][0], kb[0], sc0);
            sc0 = wmma_bf16(qa[mt][1], kb[1], sc0);
            sc1 = wmma_bf16(qa[mt][0], kb[2], sc1);
            sc1 = wmma_bf16(qa[mt][1], kb[3], sc1);

            // ---- causal mask + online softmax ----
            float p0[8], p1[8], corr[8];
#pragma unroll
            for (int v = 0; v < 8; ++v) {
                const int rowg = q0 + mt * 16 + v + 8 * half;
                float a0 = sc0[v] * scale;
                float a1 = sc1[v] * scale;
                if (s0 + col > rowg)      a0 = -3.0e38f;
                if (s0 + 16 + col > rowg) a1 = -3.0e38f;
                float mx = fmaxf(a0, a1);
#pragma unroll
                for (int off = 1; off < 16; off <<= 1)
                    mx = fmaxf(mx, __shfl_xor(mx, off, 16));
                const float mn = fmaxf(m[mt][v], mx);
                const float cr = __expf(m[mt][v] - mn);
                const float e0 = __expf(a0 - mn);
                const float e1 = __expf(a1 - mn);
                float rs = e0 + e1;
#pragma unroll
                for (int off = 1; off < 16; off <<= 1)
                    rs += __shfl_xor(rs, off, 16);
                l[mt][v] = l[mt][v] * cr + rs;
                m[mt][v] = mn;
                corr[v]  = cr;
                p0[v]    = e0;
                p1[v]    = e1;
            }
#pragma unroll
            for (int n = 0; n < 4; ++n)
#pragma unroll
                for (int v = 0; v < 8; ++v)
                    oc[mt][n][v] *= corr[v];

            // ---- stage P (16x32 bf16) through LDS: C-frag -> A-frag ----
#pragma unroll
            for (int v = 0; v < 8; ++v) {
                const int r = v + 8 * half;
                pbuf[wid][mt][r * 32 + col]      = f32_to_bf16(p0[v]);
                pbuf[wid][mt][r * 32 + 16 + col] = f32_to_bf16(p1[v]);
            }
        }
        asm volatile("s_wait_dscnt 0" ::: "memory");  // intra-wave DS RAW

        v16bf pa[2];
        pa[0] = load_frag_a(&pbuf[wid][0][0], 32);
        pa[1] = load_frag_a(&pbuf[wid][1][0], 32);

        // ---- PV: V^T fragments are contiguous (b128) and shared by M tiles
#pragma unroll
        for (int n = 0; n < 4; ++n) {
            v16bf vb = load_frag_b_rowN(vtp + (size_t)(n * 16) * 2048 + s0, 2048);
            oc[0][n] = wmma_bf16(pa[0], vb, oc[0][n]);
            oc[1][n] = wmma_bf16(pa[1], vb, oc[1][n]);
        }
    }

    // ---- normalize and store bf16 into [B,T,C] ----
#pragma unroll
    for (int mt = 0; mt < 2; ++mt)
#pragma unroll
        for (int v = 0; v < 8; ++v) {
            const int   t   = q0 + mt * 16 + v + 8 * half;
            const float inv = 1.0f / l[mt][v];
#pragma unroll
            for (int n = 0; n < 4; ++n) {
                const int d = n * 16 + col;
                ao[((size_t)b_ * 2048 + t) * 1024 + h * 64 + d] =
                    f32_to_bf16(oc[mt][n][v] * inv);
            }
        }
}

// ---------------------------------------------------------------------------
// Kernel 4: y = attn_out @ w_out^T -> fp32 [4096,1024], 64x64 blocked.
// ---------------------------------------------------------------------------
__global__ __launch_bounds__(256)
void gemm_out(const unsigned short* __restrict__ ab,
              const unsigned short* __restrict__ wb,
              float* __restrict__ out) {
    const int NMAC = 1024 / 64;                    // 16 macro N tiles
    const int wave = (blockIdx.x * blockDim.x + threadIdx.x) >> 5;
    const int m0   = (wave / NMAC) * 64;
    const int n0   = (wave % NMAC) * 64;

    v8f c[4][4] = {};
    for (int k = 0; k < 1024; k += 32) {
        v16bf a[4], b[4];
#pragma unroll
        for (int i = 0; i < 4; ++i)
            a[i] = load_frag_a(ab + (size_t)(m0 + i * 16) * 1024 + k, 1024);
#pragma unroll
        for (int j = 0; j < 4; ++j)
            b[j] = load_frag_b_rowN(wb + (size_t)(n0 + j * 16) * 1024 + k, 1024);
#pragma unroll
        for (int i = 0; i < 4; ++i)
#pragma unroll
            for (int j = 0; j < 4; ++j)
                c[i][j] = wmma_bf16(a[i], b[j], c[i][j]);
    }

    const int lane = threadIdx.x & 31;
    const int col  = lane & 15;
    const int half = lane >> 4;
#pragma unroll
    for (int i = 0; i < 4; ++i)
#pragma unroll
        for (int j = 0; j < 4; ++j)
#pragma unroll
            for (int v = 0; v < 8; ++v) {
                const int row = m0 + i * 16 + v + 8 * half;
                out[(size_t)row * 1024 + n0 + j * 16 + col] = c[i][j][v];
            }
}

// ---------------------------------------------------------------------------
// Host entry
// ---------------------------------------------------------------------------
extern "C" void kernel_launch(void* const* d_in, const int* in_sizes, int n_in,
                              void* d_out, int out_size, void* d_ws, size_t ws_size,
                              hipStream_t stream) {
    const float* x    = (const float*)d_in[0];   // [2,2048,1024]
    const float* wqkv = (const float*)d_in[1];   // [3072,1024]
    const float* wout = (const float*)d_in[2];   // [1024,1024]
    float*       out  = (float*)d_out;           // [2,2048,1024]

    unsigned short* ws    = (unsigned short*)d_ws;
    unsigned short* xb    = ws;                              // 4096*1024
    unsigned short* wqkvb = xb + (size_t)4096 * 1024;        // 3072*1024
    unsigned short* woutb = wqkvb + (size_t)3072 * 1024;     // 1024*1024
    unsigned short* qkb   = woutb + (size_t)1024 * 1024;     // 2*2*16*2048*64
    unsigned short* vtb   = qkb + (size_t)2 * 2 * 16 * 2048 * 64;   // 2*16*64*2048
    unsigned short* aob   = vtb + (size_t)2 * 16 * 64 * 2048;       // 4096*1024

    const int nX = 2 * 2048 * 1024;
    const int nW = 3072 * 1024;
    const int nO = 1024 * 1024;
    cvt_f32_bf16<<<(nX + 255) / 256, 256, 0, stream>>>(x, xb, nX);
    cvt_f32_bf16<<<(nW + 255) / 256, 256, 0, stream>>>(wqkv, wqkvb, nW);
    cvt_f32_bf16<<<(nO + 255) / 256, 256, 0, stream>>>(wout, woutb, nO);

    // 64 macro-M x 48 macro-N waves, 8 waves per 256-thread block
    gemm_qkv<<<(64 * 48) / 8, 256, 0, stream>>>(xb, wqkvb, qkb, vtb);

    // 32 (b,h) * 64 query tiles (32 rows each) = 2048 waves
    flash_attn<<<2048 / 8, 256, 0, stream>>>(qkb, vtb, aob);

    // 64 macro-M x 16 macro-N waves
    gemm_out<<<(64 * 16) / 8, 256, 0, stream>>>(aob, woutb, out);
}
